// CRSDSequence_8031588843737
// MI455X (gfx1250) — matrix operand order repaired and checked by reading, not compiled, not run on hardware
//
#include <hip/hip_runtime.h>
#include <stdint.h>

// ---------------- problem constants ----------------
#define V_  32000
#define E_  512
#define H_  1024
#define R1_ 512
#define R2_ 512
#define B_  8
#define T_  512
#define MPAD 16               // batch padded to WMMA M=16
#define NWG_REC 4             // workgroups in persistent recurrence kernel

typedef __attribute__((ext_vector_type(8)))  __bf16 v8bf;
typedef __attribute__((ext_vector_type(16))) __bf16 v16bf;
typedef __attribute__((ext_vector_type(8)))  float  v8f;
typedef int v4i_vs __attribute__((__vector_size__(16)));   // matches builtin param

// async global->LDS path (gfx1250): guarded so either toolchain compiles
#if defined(__AMDGCN__) && __has_builtin(__builtin_amdgcn_global_load_async_to_lds_b128) && __has_builtin(__builtin_amdgcn_s_wait_asynccnt)
#define USE_ASYNC_LDS 1
#else
#define USE_ASYNC_LDS 0
#endif

// ---------------- ws layout (bytes) ----------------
#define OFF_BAR    ((size_t)0)                       // 256 B (barrier counters)
#define OFF_XB     ((size_t)256)                     // [T,16,E] bf16
#define OFF_WIN1   (OFF_XB    + (size_t)T_*MPAD*E_*2)
#define OFF_WRES1  (OFF_WIN1  + (size_t)E_*R1_*2)
#define OFF_WIN2   (OFF_WRES1 + (size_t)R1_*R1_*2)
#define OFF_WRES2  (OFF_WIN2  + (size_t)R1_*R2_*2)
#define OFF_WXH    (OFF_WRES2 + (size_t)R2_*R2_*2)
#define OFF_WHH    (OFF_WXH   + (size_t)E_*H_*2)
#define OFF_WRH    (OFF_WHH   + (size_t)H_*H_*2)
#define OFF_WOUT   (OFF_WRH   + (size_t)R2_*H_*2)
#define OFF_R1S    (OFF_WOUT  + (size_t)H_*V_*2)     // 2 x [16,R1] bf16 (ping-pong)
#define OFF_R2S    (OFF_R1S   + (size_t)2*MPAD*R1_*2)
#define OFF_HS     (OFF_R2S   + (size_t)2*MPAD*R2_*2) // [T+1,16,H] bf16

// ---------------- helpers ----------------
static __device__ __forceinline__ float fast_tanh(float x) {
#if __has_builtin(__builtin_amdgcn_tanhf)
  return __builtin_amdgcn_tanhf(x);   // V_TANH_F32 (gfx1250 TRANS op)
#else
  return tanhf(x);
#endif
}

static __device__ __forceinline__ v8f wmma_bf16(v16bf a, v16bf b, v8f c) {
  // v_wmma_f32_16x16x32_bf16
  return __builtin_amdgcn_wmma_f32_16x16x32_bf16(false, a, false, b, (short)0, c,
                                                 false, false);
}

// A fragment from row-major bf16 [16, ld], K-slice starting at kbase.
// Lane l: row = l&15; lanes 0-15 hold K kbase+[0..15], lanes 16-31 +16.
static __device__ __forceinline__ v16bf load_a_frag(const __bf16* a,
                                                    int ld, int kbase, int lane) {
  const int row = lane & 15;
  const int ko  = kbase + (lane & 16);          // +16 for upper half-wave
  const __bf16* p = a + (size_t)row * ld + ko;
  v8bf lo = *(const v8bf*)(p);
  v8bf hi = *(const v8bf*)(p + 8);
  return __builtin_shufflevector(lo, hi, 0, 1, 2, 3, 4, 5, 6, 7,
                                 8, 9, 10, 11, 12, 13, 14, 15);
}

// Accumulate A[16,K] (row-major bf16, ld) x packed-W tile column nt into acc,
// with a depth-2 software pipeline: operands for k+1/k+2 are in flight while
// WMMA k executes (hides L2 latency in the serial recurrence).
// NOTE: no __restrict__ here or at callers — the possible aliasing with the
// state stores keeps these loads streaming per time-step (L2-resident) instead
// of being LICM-hoisted and spilled to scratch.
static __device__ __forceinline__ void mma_loop(v8f& acc, const __bf16* arow,
                                                int ld, const __bf16* wp,
                                                int Kt, int nt, int lane) {
  const __bf16* tp = wp + ((size_t)nt * Kt) * 512 + lane * 16;
  v16bf a0 = load_a_frag(arow, ld, 0, lane);
  v16bf b0 = *(const v16bf*)(tp);
  v16bf a1 = load_a_frag(arow, ld, 32, lane);
  v16bf b1 = *(const v16bf*)(tp + 512);
  for (int kt = 0; kt < Kt; ++kt) {
    v16bf a2 = {}, b2 = {};
    if (kt + 2 < Kt) {
      a2 = load_a_frag(arow, ld, (kt + 2) * 32, lane);
      b2 = *(const v16bf*)(tp + (size_t)(kt + 2) * 512);
    }
    acc = wmma_bf16(a0, b0, acc);
    a0 = a1; b0 = b1;                 // rotation vanishes after full unroll
    a1 = a2; b1 = b2;
  }
}

// Agent-scope barrier for the persistent recurrence kernel.
static __device__ __forceinline__ void grid_barrier(unsigned* bar, unsigned* phase) {
  __syncthreads();
  if (threadIdx.x == 0) {
    const unsigned ph = *phase;
    __threadfence();
    unsigned arrived = __hip_atomic_fetch_add(&bar[0], 1u, __ATOMIC_ACQ_REL,
                                              __HIP_MEMORY_SCOPE_AGENT) + 1u;
    if (arrived == (ph + 1u) * NWG_REC) {
      __hip_atomic_store(&bar[1], ph + 1u, __ATOMIC_RELEASE,
                         __HIP_MEMORY_SCOPE_AGENT);
    } else {
      while (__hip_atomic_load(&bar[1], __ATOMIC_ACQUIRE,
                               __HIP_MEMORY_SCOPE_AGENT) < ph + 1u) {
        __builtin_amdgcn_s_sleep(1);
      }
    }
  }
  __syncthreads();
  *phase += 1u;
}

// ---------------- kernels ----------------
__global__ void zero_kernel(unsigned* p, size_t n) {
  for (size_t i = (size_t)blockIdx.x * blockDim.x + threadIdx.x; i < n;
       i += (size_t)gridDim.x * blockDim.x)
    p[i] = 0u;
}

// gather + fp32->bf16, rows b>=8 zero-padded: Xb[t][b][e]
__global__ void embed_kernel(const int* __restrict__ ids, const float* __restrict__ emb,
                             __bf16* __restrict__ xb) {
  const size_t total = (size_t)T_ * MPAD * E_;
  for (size_t idx = (size_t)blockIdx.x * blockDim.x + threadIdx.x; idx < total;
       idx += (size_t)gridDim.x * blockDim.x) {
    int t = (int)(idx / (MPAD * E_));
    int r = (int)(idx % (MPAD * E_));
    int b = r / E_;
    int e = r % E_;
    float v = 0.0f;
    if (b < B_) v = emb[(size_t)ids[b * T_ + t] * E_ + e];
    xb[idx] = (__bf16)v;
  }
}

// Pack fp32 W[K,N] row-major into bf16 WMMA B-operand tiles (32x16),
// tile order (nt, kt): k-tiles contiguous per column block.
__global__ void pack_weight_kernel(const float* __restrict__ w, __bf16* __restrict__ p,
                                   int K, int N) {
  const int Kt = K >> 5;
  const size_t total = (size_t)K * N;
  for (size_t idx = (size_t)blockIdx.x * blockDim.x + threadIdx.x; idx < total;
       idx += (size_t)gridDim.x * blockDim.x) {
    size_t tile = idx >> 9;
    int r = (int)(idx & 511);
    int l = r >> 4;
    int q = r & 15;
    int nt = (int)(tile / Kt);
    int kt = (int)(tile % Kt);
    int k = kt * 32 + ((l >> 4) << 4) + q;   // lanes 16-31 hold K=16..31
    int n = nt * 16 + (l & 15);
    p[idx] = (__bf16)w[(size_t)k * N + n];
  }
}

// W_out pack: strip layout. Tile index = (ns*Kt + kt)*4 + j so the 4 tiles of a
// 64-column strip at one k-step are a contiguous 4KB block (one async stage).
__global__ void pack_wout_kernel(const float* __restrict__ w, __bf16* __restrict__ p) {
  const int Kt = H_ >> 5;                 // 32
  const size_t total = (size_t)H_ * V_;
  for (size_t idx = (size_t)blockIdx.x * blockDim.x + threadIdx.x; idx < total;
       idx += (size_t)gridDim.x * blockDim.x) {
    size_t tile = idx >> 9;
    int r = (int)(idx & 511);
    int l = r >> 4;
    int q = r & 15;
    int j  = (int)(tile & 3);
    int sk = (int)(tile >> 2);            // = ns*Kt + kt
    int ns = sk / Kt;
    int kt = sk % Kt;
    int nt = ns * 4 + j;
    int k = kt * 32 + ((l >> 4) << 4) + q;
    int n = nt * 16 + (l & 15);
    p[idx] = (__bf16)w[(size_t)k * V_ + n];
  }
}

static __device__ __forceinline__ void blend_store(v8f acc, const __bf16* oldv,
                                                   __bf16* newv, int ld, int nt,
                                                   int lane) {
  union { v8f v; float f[8]; } u; u.v = acc;
  const int n = nt * 16 + (lane & 15);
  const int mbase = (lane & 16) >> 1;
  #pragma unroll
  for (int i = 0; i < 8; ++i) {
    const int m = mbase + i;
    float ov = (float)oldv[(size_t)m * ld + n];
    float y = 0.5f * ov + 0.5f * fast_tanh(u.f[i]);
    newv[(size_t)m * ld + n] = (__bf16)y;
  }
}

static __device__ __forceinline__ void tanh_store(v8f acc, const float* bias,
                                                  __bf16* newv, int ld, int nt,
                                                  int lane) {
  union { v8f v; float f[8]; } u; u.v = acc;
  const int n = nt * 16 + (lane & 15);
  const float bn = bias[n];
  const int mbase = (lane & 16) >> 1;
  #pragma unroll
  for (int i = 0; i < 8; ++i) {
    const int m = mbase + i;
    newv[(size_t)m * ld + n] = (__bf16)fast_tanh(u.f[i] + bn);
  }
}

// Persistent kernel: whole T=512 recurrence on-device, 4 WGs x 8 waves.
// Pointers intentionally NOT __restrict__ (see mma_loop note).
__global__ void recurrence_kernel(const __bf16* xb, const __bf16* win1,
                                  const __bf16* wres1, const __bf16* win2,
                                  const __bf16* wres2, const __bf16* wxh,
                                  const __bf16* whh, const __bf16* wrh,
                                  const float* b_h, __bf16* r1s,
                                  __bf16* r2s, __bf16* hs,
                                  unsigned* bar) {
  const int lane = threadIdx.x & 31;
  const int wave = (blockIdx.x * blockDim.x + threadIdx.x) >> 5;  // 0..31
  unsigned phase = 0;

  for (int t = 0; t < T_; ++t) {
    asm volatile("" ::: "memory");   // no cross-step hoisting of weight tiles
    const int p = t & 1;
    const __bf16* xrow  = xb + (size_t)t * MPAD * E_;
    const __bf16* r1old = r1s + (size_t)p * MPAD * R1_;
    __bf16*       r1new = r1s + (size_t)(1 - p) * MPAD * R1_;
    const __bf16* r2old = r2s + (size_t)p * MPAD * R2_;
    __bf16*       r2new = r2s + (size_t)(1 - p) * MPAD * R2_;
    const __bf16* hold  = hs + (size_t)t * MPAD * H_;
    __bf16*       hnew  = hs + (size_t)(t + 1) * MPAD * H_;

    // Stage A: r1 <- 0.5*r1 + 0.5*tanh(x@W_in1 + r1@W_res1); 32 tiles, 1/wave
    {
      const int nt = wave;
      v8f acc = {};
      mma_loop(acc, xrow,  E_,  win1,  E_ / 32,  nt, lane);
      mma_loop(acc, r1old, R1_, wres1, R1_ / 32, nt, lane);
      blend_store(acc, r1old, r1new, R1_, nt, lane);
    }
    grid_barrier(bar, &phase);

    // Stage B: r2 update
    {
      const int nt = wave;
      v8f acc = {};
      mma_loop(acc, r1new, R1_, win2,  R1_ / 32, nt, lane);
      mma_loop(acc, r2old, R2_, wres2, R2_ / 32, nt, lane);
      blend_store(acc, r2old, r2new, R2_, nt, lane);
    }
    grid_barrier(bar, &phase);

    // Stage C: h <- tanh(x@W_xh + h@W_hh + r2@W_rh + b_h); 64 tiles, 2/wave
    #pragma unroll
    for (int j = 0; j < 2; ++j) {
      const int nt = wave * 2 + j;
      v8f acc = {};
      mma_loop(acc, xrow,  E_,  wxh, E_ / 32,  nt, lane);
      mma_loop(acc, hold,  H_,  whh, H_ / 32,  nt, lane);
      mma_loop(acc, r2new, R2_, wrh, R2_ / 32, nt, lane);
      tanh_store(acc, b_h, hnew, H_, nt, lane);
    }
    grid_barrier(bar, &phase);
  }
}

// Stage one 4KB strip block (4 B-tiles at k-step kt) into LDS; 256 lanes x 16B.
static __device__ __forceinline__ void stage_strip(const __bf16* wout,
                                                   __bf16* lds, int sk, int tid) {
  const __bf16* g = wout + (size_t)sk * 2048 + tid * 8;
  __bf16* l = lds + tid * 8;
#if USE_ASYNC_LDS
  __builtin_amdgcn_global_load_async_to_lds_b128(
      (__attribute__((address_space(1))) v4i_vs*)g,
      (__attribute__((address_space(3))) v4i_vs*)l, 0, 0);
#else
  *(v8bf*)l = *(const v8bf*)g;
#endif
}

// Projection: WG = 8 waves share one 64-col strip of W_out (staged via async
// copies to LDS, double buffered), each wave owns one t (16-row A block).
__global__ void proj_kernel(const __bf16* __restrict__ hs, const __bf16* __restrict__ wout,
                            const float* __restrict__ b_out, float* __restrict__ out) {
  __shared__ __align__(16) __bf16 bsh[2][2048];   // 2 x 4KB
  const int tid  = threadIdx.x;
  const int lane = tid & 31;
  const int wv   = tid >> 5;                      // 0..7
  const int nstrips = V_ / 64;                    // 500
  const int tgroup = blockIdx.x / nstrips;
  const int ns     = blockIdx.x % nstrips;
  const int t = tgroup * 8 + wv;
  const __bf16* arow = hs + (size_t)(t + 1) * MPAD * H_;
  const int Kt = H_ / 32;                         // 32

  union { v8f v; float f[8]; } acc[4];
  const int ncol = lane & 15;
  #pragma unroll
  for (int j = 0; j < 4; ++j) {
    const float bo = b_out[ns * 64 + j * 16 + ncol];
    #pragma unroll
    for (int i = 0; i < 8; ++i) acc[j].f[i] = bo;
  }

  stage_strip(wout, bsh[0], ns * Kt + 0, tid);    // prologue: kt=0 -> buf 0

  for (int kt = 0; kt < Kt; ++kt) {
    const int p = kt & 1;
    // A fragment load issued early: its global latency hides behind the
    // async-copy wait + block barrier below.
    v16bf a = load_a_frag(arow, H_, kt * 32, lane);
    if (kt + 1 < Kt) {
      stage_strip(wout, bsh[p ^ 1], ns * Kt + (kt + 1), tid);
#if USE_ASYNC_LDS
      __builtin_amdgcn_s_wait_asynccnt(1);        // current buffer's copy done
#endif
    } else {
#if USE_ASYNC_LDS
      __builtin_amdgcn_s_wait_asynccnt(0);
#endif
    }
    __syncthreads();                              // buf p fully written by all waves

    #pragma unroll
    for (int j = 0; j < 4; ++j) {
      v16bf b = *(const v16bf*)(&bsh[p][j * 512 + lane * 16]);
      acc[j].v = wmma_bf16(a, b, acc[j].v);
    }
    __syncthreads();                              // done reading buf p
  }

  const int mbase = (lane & 16) >> 1;
  #pragma unroll
  for (int j = 0; j < 4; ++j) {
    const int n = ns * 64 + j * 16 + ncol;
    #pragma unroll
    for (int i = 0; i < 8; ++i) {
      const int b = mbase + i;                    // batch row; >=8 is padding
      if (b < B_) out[((size_t)b * T_ + t) * V_ + n] = acc[j].f[i];
    }
  }
}

// ---------------- host launcher ----------------
extern "C" void kernel_launch(void* const* d_in, const int* in_sizes, int n_in,
                              void* d_out, int out_size, void* d_ws, size_t ws_size,
                              hipStream_t stream) {
  (void)in_sizes; (void)n_in; (void)out_size; (void)ws_size;
  const int*   ids    = (const int*)d_in[0];
  const float* emb    = (const float*)d_in[1];
  const float* W_in1  = (const float*)d_in[2];
  const float* W_res1 = (const float*)d_in[3];
  const float* W_in2  = (const float*)d_in[4];
  const float* W_res2 = (const float*)d_in[5];
  const float* W_xh   = (const float*)d_in[6];
  const float* W_hh   = (const float*)d_in[7];
  const float* W_rh   = (const float*)d_in[8];
  const float* b_h    = (const float*)d_in[9];
  const float* W_out  = (const float*)d_in[10];
  const float* b_out  = (const float*)d_in[11];
  float* out = (float*)d_out;
  char*  ws  = (char*)d_ws;

  unsigned* bar    = (unsigned*)(ws + OFF_BAR);
  __bf16*   xb     = (__bf16*)(ws + OFF_XB);
  __bf16*   pin1   = (__bf16*)(ws + OFF_WIN1);
  __bf16*   pres1  = (__bf16*)(ws + OFF_WRES1);
  __bf16*   pin2   = (__bf16*)(ws + OFF_WIN2);
  __bf16*   pres2  = (__bf16*)(ws + OFF_WRES2);
  __bf16*   pxh    = (__bf16*)(ws + OFF_WXH);
  __bf16*   phh    = (__bf16*)(ws + OFF_WHH);
  __bf16*   prh    = (__bf16*)(ws + OFF_WRH);
  __bf16*   pout   = (__bf16*)(ws + OFF_WOUT);
  __bf16*   r1s    = (__bf16*)(ws + OFF_R1S);
  __bf16*   r2s    = (__bf16*)(ws + OFF_R2S);
  __bf16*   hsbuf  = (__bf16*)(ws + OFF_HS);

  // zero: barrier words, r1/r2 ping-pong state, h[0]
  zero_kernel<<<1, 64, 0, stream>>>(bar, 64);
  zero_kernel<<<64, 256, 0, stream>>>((unsigned*)r1s,
                                      ((size_t)2 * MPAD * (R1_ + R2_) * 2) / 4);
  zero_kernel<<<32, 256, 0, stream>>>((unsigned*)hsbuf,
                                      ((size_t)MPAD * H_ * 2) / 4);

  embed_kernel<<<4096, 256, 0, stream>>>(ids, emb, xb);

  pack_weight_kernel<<<512, 256, 0, stream>>>(W_in1,  pin1,  E_,  R1_);
  pack_weight_kernel<<<512, 256, 0, stream>>>(W_res1, pres1, R1_, R1_);
  pack_weight_kernel<<<512, 256, 0, stream>>>(W_in2,  pin2,  R1_, R2_);
  pack_weight_kernel<<<512, 256, 0, stream>>>(W_res2, pres2, R2_, R2_);
  pack_weight_kernel<<<1024, 256, 0, stream>>>(W_xh,  pxh,   E_,  H_);
  pack_weight_kernel<<<2048, 256, 0, stream>>>(W_hh,  phh,   H_,  H_);
  pack_weight_kernel<<<1024, 256, 0, stream>>>(W_rh,  prh,   R2_, H_);
  pack_wout_kernel<<<8192, 256, 0, stream>>>(W_out, pout);

  recurrence_kernel<<<NWG_REC, 256, 0, stream>>>(xb, pin1, pres1, pin2, pres2,
                                                 pxh, phh, prh, b_h, r1s, r2s,
                                                 hsbuf, bar);

  // (T/8) t-groups x (V/64) strips
  proj_kernel<<<(T_ / 8) * (V_ / 64), 256, 0, stream>>>(hsbuf, pout, b_out, out);
}
